// AFWP_89438398972451
// MI455X (gfx1250) — matrix operand order repaired and checked by reading, not compiled, not run on hardware
//
#include <hip/hip_runtime.h>
#include <hip/hip_bf16.h>

// ---------------------------------------------------------------------------
// Gated linear attention (per-kd decay) for MI455X / gfx1250.
//   1) convert x / Wq / Wk / Wo to bf16
//   2) q = x@Wq^T, k = sigmoid(x@Wk^T + b)   (WMMA GEMM, async LDS-staged A
//                                             tile, ping-pong B fragments)
//   3) chunked GLA recurrence (C=16): per chunk
//        A = mask(q~ @ k~^T) (2 WMMA), h = q~@S + A@X (3), S' = Pc⊙S + K^T@X (4)
//      64 single-wave blocks; chunk staging double-buffered through the CDNA5
//      async global->LDS engine (ASYNCcnt).
//   4) out = h @ Wo^T (f32 store)
// ---------------------------------------------------------------------------

#define NB_    2
#define TT_    1024
#define HH_    8
#define DM_    512
#define NT_    (NB_ * TT_)      // 2048 rows
#define NCHUNK (TT_ / 16)       // 64 chunks of 16

typedef __bf16 bf16;
typedef __attribute__((ext_vector_type(8)))  bf16  bf16x8;
typedef __attribute__((ext_vector_type(16))) bf16  v16bf;
typedef __attribute__((ext_vector_type(8)))  float v8f;
typedef int v4i_ __attribute__((vector_size(16)));   // matches builtin proto

#if defined(__has_builtin)
#if __has_builtin(__builtin_amdgcn_global_load_async_to_lds_b128)
#define ASYNC_LDS 1
#else
#define ASYNC_LDS 0
#endif
#else
#define ASYNC_LDS 0
#endif

// 16-byte global -> LDS copy through the async DMA engine when available.
__device__ __forceinline__ void async_cp16(void* ldst, const void* gsrc) {
#if ASYNC_LDS
  void* g = const_cast<void*>(gsrc);
  __builtin_amdgcn_global_load_async_to_lds_b128(
      (__attribute__((address_space(1))) v4i_*)g,
      (__attribute__((address_space(3))) v4i_*)ldst, 0, 0);
#else
  *(bf16x8*)ldst = *(const bf16x8*)gsrc;
#endif
}

__device__ __forceinline__ void async_wait() {
#if ASYNC_LDS
#if __has_builtin(__builtin_amdgcn_s_wait_asynccnt)
  __builtin_amdgcn_s_wait_asynccnt(0);
#else
  asm volatile("s_wait_asynccnt 0x0" ::: "memory");
#endif
#endif
}

__device__ __forceinline__ v8f zero8() {
  v8f z;
#pragma unroll
  for (int i = 0; i < 8; ++i) z[i] = 0.f;
  return z;
}

// A-matrix fragment (16x32 bf16, MxK): lane<16 holds K {kb..kb+7, kb+16..kb+23},
// lane>=16 holds K {kb+8..kb+15, kb+24..kb+31}  (ISA 7.12.2 16-bit A layout)
__device__ __forceinline__ v16bf fragA(const bf16* row, int kbase, int sel) {
  const bf16x8* p = (const bf16x8*)(row + kbase);
  bf16x8 c0 = p[sel];
  bf16x8 c1 = p[2 + sel];
  v16bf r;
#pragma unroll
  for (int i = 0; i < 8; ++i) { r[i] = c0[i]; r[8 + i] = c1[i]; }
  return r;
}

// B-matrix fragment (32x16 bf16, KxN), lane holds column N=lane%16:
// lane<16 -> K {kb..kb+7, kb+16..kb+23}, lane>=16 -> K {kb+8..kb+15, kb+24..kb+31} (matches fragA)
__device__ __forceinline__ v16bf fragB(const bf16* row, int kbase, int sel) {
  const bf16x8* p = (const bf16x8*)(row + kbase);
  bf16x8 c0 = p[sel];
  bf16x8 c1 = p[2 + sel];
  v16bf r;
#pragma unroll
  for (int i = 0; i < 8; ++i) { r[i] = c0[i]; r[8 + i] = c1[i]; }
  return r;
}

__device__ __forceinline__ v8f wmma_bf16(v16bf a, v16bf b, v8f c) {
  return __builtin_amdgcn_wmma_f32_16x16x32_bf16(false, a, false, b, (short)0, c,
                                                 false, false);
}

// ---------------------------------------------------------------------------
__global__ void cvt_f32_bf16(const float* __restrict__ in, bf16* __restrict__ out,
                             int n) {
  int i = blockIdx.x * blockDim.x + threadIdx.x;
  if (i < n) out[i] = (bf16)in[i];
}

// ---------------------------------------------------------------------------
// C[M=2048][512] = A[2048][512] @ W[512][512]^T   (bf16 in, WMMA, f32 acc)
// Block = one 16-row A tile (async-staged to LDS, shared by 8 waves covering
// all 512 output cols). B fragments ping-pong buffered from global; the k-loop
// is unrolled by 2 so buffer indices are compile-time constant (no reg copies).
// mode 0: store bf16   mode 1: sigmoid(x+bias) -> bf16   mode 2: store f32
__global__ void __launch_bounds__(256)
gemm_bf16(const bf16* __restrict__ A, const bf16* __restrict__ W,
          const float* __restrict__ bias, void* __restrict__ outp, int mode) {
  __shared__ __align__(16) bf16 sA[16][DM_];            // 16 KB
  const int tid = threadIdx.x;
  const int lane = tid & 31, sel = lane >> 4, l16 = lane & 15;
  const int tileNw = tid >> 5;                          // wave id = N panel
  const int tileM = blockIdx.x;

  // async-stage the 16x512 A tile: 1024 x 16B, 4 per thread
  for (int c = tid; c < 16 * (DM_ / 8); c += 256) {
    const int r = c >> 6, k8 = c & 63;
    async_cp16(&sA[r][k8 * 8], A + (size_t)(tileM * 16 + r) * DM_ + k8 * 8);
  }
  async_wait();
  __syncthreads();

  const bf16* wbase = W + (size_t)(tileNw * 64 + l16) * DM_;
  __builtin_prefetch(wbase, 0, 1);                      // global_prefetch

  v8f acc[4];
#pragma unroll
  for (int n = 0; n < 4; ++n) acc[n] = zero8();

  v16bf aB[2];
  v16bf bB[2][4];
  aB[0] = fragA(sA[l16], 0, sel);
#pragma unroll
  for (int n = 0; n < 4; ++n) bB[0][n] = fragB(wbase + (size_t)(n * 16) * DM_, 0, sel);

#pragma unroll 2
  for (int ks = 0; ks < DM_ / 32; ++ks) {
    const int cur = ks & 1, nxt = cur ^ 1;
    if (ks + 1 < DM_ / 32) {
      const int kn = (ks + 1) * 32;
      aB[nxt] = fragA(sA[l16], kn, sel);
#pragma unroll
      for (int n = 0; n < 4; ++n)
        bB[nxt][n] = fragB(wbase + (size_t)(n * 16) * DM_, kn, sel);
    }
#pragma unroll
    for (int n = 0; n < 4; ++n) acc[n] = wmma_bf16(aB[cur], bB[cur][n], acc[n]);
  }

#pragma unroll
  for (int n = 0; n < 4; ++n) {
    const int col = tileNw * 64 + n * 16 + l16;
#pragma unroll
    for (int r = 0; r < 8; ++r) {
      const int row = tileM * 16 + r + 8 * sel;
      float v = acc[n][r];
      if (mode == 1) { v += bias[col]; v = 1.f / (1.f + __expf(-v)); }
      if (mode == 2) ((float*)outp)[(size_t)row * DM_ + col] = v;
      else           ((bf16*)outp)[(size_t)row * DM_ + col] = (bf16)v;
    }
  }
}

// ---------------------------------------------------------------------------
// Chunked gated linear recurrence. 64 blocks x 1 wave:
// block b -> (batch,head) = b>>2, vd-tile = (b&3)*16. State slice S is
// 64(kd) x 16(vd) in 4 x v8f accumulators; bf16 copy in LDS for q~@S.
// Chunk inputs (q,k,x-slice) double-buffered via async global->LDS DMA.
__global__ void __launch_bounds__(32)
glr_kernel(const bf16* __restrict__ qg, const bf16* __restrict__ kg,
           const bf16* __restrict__ xg, bf16* __restrict__ hg) {
  __shared__ __align__(16) bf16  sKraw[2][16][64];  // async staging (k)
  __shared__ __align__(16) bf16  sQraw[2][16][64];  // async staging (q)
  __shared__ __align__(16) bf16  sXraw[2][16][16];  // async staging (x slice)
  __shared__ __align__(16) float sKf[16][64];       // k chunk (f32)
  __shared__ __align__(16) float sP [16][64];       // inclusive decay cumprod
  __shared__ __align__(16) bf16  sQt[16][64];       // q~ = q * P
  __shared__ __align__(16) bf16  sKt[16][64];       // k~ = k / P
  __shared__ __align__(16) bf16  sKhT[64][32];      // K^T: [kd][s], s>=16 pad 0
  __shared__ __align__(16) bf16  sXT[16][32];       // X^T slice: [vd][s], pad
  __shared__ __align__(16) bf16  sA [16][32];       // masked A: [t][s], pad
  __shared__ __align__(16) bf16  sST[16][64];       // S^T: [vd][kd]

  const int lane = threadIdx.x & 31, sel = lane >> 4, l16 = lane & 15;
  const int b  = blockIdx.x;
  const int nh = b >> 2, vt = b & 3;
  const int nb = nh >> 3, h = nh & 7;
  const int v0 = vt * 16;
  const size_t base = (size_t)nb * TT_ * DM_ + h * 64;

  // zero state + K-pad columns
  for (int e = lane; e < 16 * 64; e += 32) sST[e >> 6][e & 63] = (bf16)0.f;
  for (int e = lane; e < 16 * 16; e += 32) {
    sXT[e >> 4][16 + (e & 15)] = (bf16)0.f;
    sA [e >> 4][16 + (e & 15)] = (bf16)0.f;
  }
  for (int e = lane; e < 64 * 16; e += 32) sKhT[e >> 4][16 + (e & 15)] = (bf16)0.f;
  v8f S[4];
#pragma unroll
  for (int kt = 0; kt < 4; ++kt) S[kt] = zero8();
  __syncthreads();

  // issue async stage of chunk `ch` into parity buffer pb
  auto stage = [&](int ch, int pb) {
    const int t0s = ch * 16;
#pragma unroll
    for (int c = 0; c < 4; ++c) {
      const int cv = lane + c * 32;            // 128 x 16B per tensor
      const int t = cv >> 3, k8 = cv & 7;
      const size_t g = base + (size_t)(t0s + t) * DM_ + k8 * 8;
      async_cp16(&sKraw[pb][t][k8 * 8], kg + g);
      async_cp16(&sQraw[pb][t][k8 * 8], qg + g);
    }
    { const int t = lane >> 1, hf = lane & 1;  // 32 x 16B (x slice)
      async_cp16(&sXraw[pb][t][hf * 8],
                 xg + base + (size_t)(t0s + t) * DM_ + v0 + hf * 8); }
  };

  stage(0, 0);

  for (int ch = 0; ch < NCHUNK; ++ch) {
    const int t0 = ch * 16;
    const int pb = ch & 1;
    async_wait();                 // drain DMA for this chunk
    __syncthreads();
    if (ch + 1 < NCHUNK) stage(ch + 1, pb ^ 1);   // overlap next DMA w/ compute

    // ---- k chunk -> f32 ----
#pragma unroll
    for (int c = 0; c < 4; ++c) {
      const int cv = lane + c * 32;
      const int t = cv >> 3, k8 = cv & 7;
      bf16x8 v = *(const bf16x8*)&sKraw[pb][t][k8 * 8];
#pragma unroll
      for (int i = 0; i < 8; ++i) sKf[t][k8 * 8 + i] = (float)v[i];
    }
    __syncthreads();

    // ---- inclusive decay cumprod per kd ----
#pragma unroll
    for (int j = 0; j < 2; ++j) {
      const int kd = lane + j * 32;
      float P = 1.f;
      for (int t = 0; t < 16; ++t) { P *= (1.f - sKf[t][kd]); sP[t][kd] = P; }
    }
    __syncthreads();

    // ---- q~, k~, K^T tiles ----
#pragma unroll
    for (int c = 0; c < 4; ++c) {
      const int cv = lane + c * 32;
      const int t = cv >> 3, k8 = cv & 7;
      bf16x8 qv = *(const bf16x8*)&sQraw[pb][t][k8 * 8];
#pragma unroll
      for (int i = 0; i < 8; ++i) {
        const int kd = k8 * 8 + i;
        const float P = sP[t][kd];
        const float ktl = sKf[t][kd] / P;
        sQt[t][kd]  = (bf16)((float)qv[i] * P);
        sKt[t][kd]  = (bf16)ktl;
        sKhT[kd][t] = (bf16)(ktl * sP[15][kd]);
      }
    }
    // ---- X slice, transposed to [vd][s] ----
#pragma unroll
    for (int i = 0; i < 8; ++i) {
      const int e = lane * 8 + i;
      const int t = e >> 4, vl = e & 15;
      sXT[vl][t] = sXraw[pb][t][vl];
    }
    __syncthreads();

    // ---- A = q~ @ k~^T  (K = kd = 64), causal mask ----
    v8f accA = zero8();
    accA = wmma_bf16(fragA(sQt[l16], 0,  sel), fragB(sKt[l16], 0,  sel), accA);
    accA = wmma_bf16(fragA(sQt[l16], 32, sel), fragB(sKt[l16], 32, sel), accA);
#pragma unroll
    for (int r = 0; r < 8; ++r) {
      const int t = r + 8 * sel, s = l16;
      sA[t][s] = (s <= t) ? (bf16)accA[r] : (bf16)0.f;
    }
    __syncthreads();

    // ---- h = q~ @ S  +  A @ X ----
    v8f hacc = zero8();
    hacc = wmma_bf16(fragA(sQt[l16], 0,  sel), fragB(sST[l16], 0,  sel), hacc);
    hacc = wmma_bf16(fragA(sQt[l16], 32, sel), fragB(sST[l16], 32, sel), hacc);
    hacc = wmma_bf16(fragA(sA [l16], 0,  sel), fragB(sXT[l16], 0,  sel), hacc);
#pragma unroll
    for (int r = 0; r < 8; ++r) {
      const int t = t0 + r + 8 * sel;
      hg[base + (size_t)t * DM_ + v0 + l16] = (bf16)hacc[r];
    }

    // ---- state update: S = P_C ⊙ S + K^T @ X ----
#pragma unroll
    for (int kt = 0; kt < 4; ++kt) {
#pragma unroll
      for (int r = 0; r < 8; ++r) {
        const int kd = kt * 16 + r + 8 * sel;
        S[kt][r] *= sP[15][kd];
      }
      S[kt] = wmma_bf16(fragA(sKhT[kt * 16 + l16], 0, sel),
                        fragB(sXT[l16], 0, sel), S[kt]);
    }
    __syncthreads();
#pragma unroll
    for (int kt = 0; kt < 4; ++kt)
#pragma unroll
      for (int r = 0; r < 8; ++r) {
        const int kd = kt * 16 + r + 8 * sel;
        sST[l16][kd] = (bf16)S[kt][r];
      }
    __syncthreads();
  }
}

// ---------------------------------------------------------------------------
extern "C" void kernel_launch(void* const* d_in, const int* in_sizes, int n_in,
                              void* d_out, int out_size, void* d_ws, size_t ws_size,
                              hipStream_t stream) {
  (void)in_sizes; (void)n_in; (void)out_size; (void)ws_size;
  const float* x     = (const float*)d_in[0];
  const float* kw    = (const float*)d_in[1];
  const float* kbias = (const float*)d_in[2];
  const float* qw    = (const float*)d_in[3];
  const float* ow    = (const float*)d_in[4];

  char* ws = (char*)d_ws;
  size_t off = 0;
  auto alloc = [&](size_t bytes) {
    void* p = ws + off;
    off += (bytes + 255) & ~(size_t)255;
    return p;
  };
  bf16* xb  = (bf16*)alloc((size_t)NT_ * DM_ * 2);
  bf16* qwb = (bf16*)alloc((size_t)DM_ * DM_ * 2);
  bf16* kwb = (bf16*)alloc((size_t)DM_ * DM_ * 2);
  bf16* owb = (bf16*)alloc((size_t)DM_ * DM_ * 2);
  bf16* qb  = (bf16*)alloc((size_t)NT_ * DM_ * 2);
  bf16* kb  = (bf16*)alloc((size_t)NT_ * DM_ * 2);
  bf16* hb  = (bf16*)alloc((size_t)NT_ * DM_ * 2);

  cvt_f32_bf16<<<(NT_ * DM_ + 255) / 256, 256, 0, stream>>>(x,  xb,  NT_ * DM_);
  cvt_f32_bf16<<<(DM_ * DM_ + 255) / 256, 256, 0, stream>>>(qw, qwb, DM_ * DM_);
  cvt_f32_bf16<<<(DM_ * DM_ + 255) / 256, 256, 0, stream>>>(kw, kwb, DM_ * DM_);
  cvt_f32_bf16<<<(DM_ * DM_ + 255) / 256, 256, 0, stream>>>(ow, owb, DM_ * DM_);

  gemm_bf16<<<128, 256, 0, stream>>>(xb, qwb, nullptr, qb, 0);          // q
  gemm_bf16<<<128, 256, 0, stream>>>(xb, kwb, kbias,  kb, 1);           // k
  glr_kernel<<<64, 32, 0, stream>>>(qb, kb, xb, hb);                    // scan
  gemm_bf16<<<128, 256, 0, stream>>>(hb, owb, nullptr, d_out, 2);       // out
}
